// IntersampleAttention_64647847739753
// MI455X (gfx1250) — matrix operand order repaired and checked by reading, not compiled, not run on hardware
//
#include <hip/hip_runtime.h>
#include <hip/hip_bf16.h>

typedef __attribute__((ext_vector_type(16))) __bf16 v16bf;
typedef __attribute__((ext_vector_type(8)))  float  v8f;

namespace {
constexpr int kB = 16;
constexpr int kH = 8;
constexpr int kL = 128;
constexpr int kD = 64;
constexpr int kQKS = 72;    // padded LDS row stride (shorts) for Q/K/V tiles
constexpr int kPS  = 136;   // padded LDS row stride (shorts) for P tile
}

__device__ __forceinline__ unsigned short f32_to_bf16_bits(float f) {
    unsigned u = __builtin_bit_cast(unsigned, f);
    u += 0x7FFFu + ((u >> 16) & 1u);           // round-to-nearest-even
    return (unsigned short)(u >> 16);
}
__device__ __forceinline__ __bf16 bf16_from_bits(unsigned short s) {
    return __builtin_bit_cast(__bf16, s);
}

__global__ __launch_bounds__(256)
void IntersampleAttention_64647847739753_kernel(const float* __restrict__ Q,
                                                const float* __restrict__ K,
                                                const float* __restrict__ V,
                                                const int*  __restrict__ Mask,
                                                float* __restrict__ outO,
                                                float* __restrict__ outA)
{
    // 3 * 128 * 72 shorts = 55296 bytes of LDS.
    __shared__ unsigned short smem[3 * kL * kQKS];
    unsigned short* sQ = smem;                  // [128][72] bf16 bits
    unsigned short* sK = smem + kL * kQKS;      // [128][72]
    unsigned short* sP = smem;                  // [128][136] overlays sQ+sK after stage 1
    unsigned short* sV = smem + 2 * kL * kQKS;  // [128][72]

    const int bid = blockIdx.x;                 // flattened (b,h,c)
    const int b = bid / (kH * kB);
    const int h = (bid / kB) % kH;
    const int c = bid % kB;

    const float* Qt = Q + (size_t)(b * kH + h) * kL * kD;   // Q[b,h,:,:]
    const float* Kt = K + (size_t)(h * kB + c) * kL * kD;   // K[h,c,:,:]
    const float* Vt = V + (size_t)(h * kB + c) * kL * kD;   // V[h,c,:,:]
    const int*   Mt = Mask + (size_t)b * kL * kL;           // mask[b,0,0,:,:]

    const int tid  = threadIdx.x;
    const int lane = tid & 31;
    const int wave = tid >> 5;                  // 8 waves; wave w owns S rows [16w,16w+16)
    const int half = lane >> 4;                 // wave32 half-select per ISA layouts
    const int l16  = lane & 15;

    // ---- Cooperative load: global f32 -> LDS bf16 ----
    for (int idx = tid; idx < kL * kD; idx += 256) {
        const int r = idx >> 6, d = idx & 63;
        sQ[r * kQKS + d] = f32_to_bf16_bits(Qt[idx]);
        sK[r * kQKS + d] = f32_to_bf16_bits(Kt[idx]);
        sV[r * kQKS + d] = f32_to_bf16_bits(Vt[idx]);
    }
    __syncthreads();

    const int i0   = wave * 16;
    const int iRow = i0 + 8 * half;             // C/D row for VGPR r is iRow + r

    // ---- Stage 1: S = Q K^T over this wave's 16 rows ----
    // A fragment (16x32 bf16): lane m = l16; element e -> k = 16*half + e
    v16bf aQ[2];
#pragma unroll
    for (int kc = 0; kc < 2; ++kc)
#pragma unroll
        for (int e = 0; e < 16; ++e) {
            const int d = kc * 32 + 16 * half + e;
            aQ[kc][e] = bf16_from_bits(sQ[(i0 + l16) * kQKS + d]);
        }

    v8f acc[8];
#pragma unroll
    for (int jb = 0; jb < 8; ++jb) {
        v8f s = {};
#pragma unroll
        for (int kc = 0; kc < 2; ++kc) {
            // B fragment (32x16 bf16): n = l16 (column j), element e -> k = 16*half + e
            v16bf bK;
#pragma unroll
            for (int e = 0; e < 16; ++e)
                bK[e] = bf16_from_bits(sK[(jb * 16 + l16) * kQKS + kc * 32 + 16 * half + e]);
            s = __builtin_amdgcn_wmma_f32_16x16x32_bf16(false, aQ[kc], false, bK,
                                                        (short)0, s, false, false);
        }
        acc[jb] = s;
    }

    // ---- Scale + mask (mask depends only on b,i,j) ----
#pragma unroll
    for (int jb = 0; jb < 8; ++jb)
#pragma unroll
        for (int r = 0; r < 8; ++r) {
            const int i = iRow + r;
            const int j = jb * 16 + l16;
            float s = acc[jb][r] * 0.125f;      // 1/sqrt(64)
            if (Mt[i * kL + j] == 0) s = -1.0e9f;
            acc[jb][r] = s;
        }

    // ---- Row softmax, fully wave-private (rows keyed by (r,half); cols by lane&jb) ----
    float rowinv[8];
#pragma unroll
    for (int r = 0; r < 8; ++r) {
        float m = -3.0e38f;
#pragma unroll
        for (int jb = 0; jb < 8; ++jb) m = fmaxf(m, acc[jb][r]);
#pragma unroll
        for (int ofs = 1; ofs < 16; ofs <<= 1)
            m = fmaxf(m, __shfl_xor(m, ofs, 32));   // stays within 16-lane half
        float s = 0.0f;
#pragma unroll
        for (int jb = 0; jb < 8; ++jb) {
            const float e = __expf(acc[jb][r] - m);
            acc[jb][r] = e;
            s += e;
        }
#pragma unroll
        for (int ofs = 1; ofs < 16; ofs <<= 1)
            s += __shfl_xor(s, ofs, 32);
        rowinv[r] = 1.0f / s;
    }

    __syncthreads();   // all waves finished reading sQ/sK before sP overlays them

    // ---- Write attention weights + stash P (bf16) in LDS for stage 2 ----
    const size_t baseA = (size_t)bid * kL * kL;
#pragma unroll
    for (int jb = 0; jb < 8; ++jb)
#pragma unroll
        for (int r = 0; r < 8; ++r) {
            const int i = iRow + r;
            const int j = jb * 16 + l16;
            const float p = acc[jb][r] * rowinv[r];
            outA[baseA + (size_t)i * kL + j] = p;
            sP[i * kPS + j] = f32_to_bf16_bits(p);
        }

    // ---- Stage 2: O = P V (wave reads only its own P rows; per-wave LDS is in-order) ----
    v16bf aP[4];
#pragma unroll
    for (int jc = 0; jc < 4; ++jc)
#pragma unroll
        for (int e = 0; e < 16; ++e) {
            const int j = jc * 32 + 16 * half + e;
            aP[jc][e] = bf16_from_bits(sP[(i0 + l16) * kPS + j]);
        }

    const size_t baseO = (size_t)bid * kL * kD;
#pragma unroll
    for (int db = 0; db < 4; ++db) {
        v8f o = {};
#pragma unroll
        for (int jc = 0; jc < 4; ++jc) {
            v16bf bV;   // B[k=j][n=d]: n = l16 (d column), k = jc*32 + 16*half + e
#pragma unroll
            for (int e = 0; e < 16; ++e)
                bV[e] = bf16_from_bits(sV[(jc * 32 + 16 * half + e) * kQKS + db * 16 + l16]);
            o = __builtin_amdgcn_wmma_f32_16x16x32_bf16(false, aP[jc], false, bV,
                                                        (short)0, o, false, false);
        }
#pragma unroll
        for (int r = 0; r < 8; ++r) {
            const int i = iRow + r;
            const int d = db * 16 + l16;
            outO[baseO + (size_t)i * kD + d] = o[r];
        }
    }
}

extern "C" void kernel_launch(void* const* d_in, const int* in_sizes, int n_in,
                              void* d_out, int out_size, void* d_ws, size_t ws_size,
                              hipStream_t stream) {
    const float* Q    = (const float*)d_in[0];
    const float* K    = (const float*)d_in[1];
    const float* V    = (const float*)d_in[2];
    const int*   mask = (const int*)d_in[3];

    float* outO = (float*)d_out;                                   // (B,H,B,L,D) f32
    float* outA = outO + (size_t)kB * kH * kB * kL * kD;           // (B,H,B,L,L) f32

    dim3 grid(kB * kH * kB);   // 2048 workgroups, one per (b,h,c)
    dim3 block(256);           // 8 waves (wave32)
    IntersampleAttention_64647847739753_kernel<<<grid, block, 0, stream>>>(
        Q, K, V, mask, outO, outA);
}
